// SingleForgetGateTreeLSTM_24739011625784
// MI455X (gfx1250) — matrix operand order repaired and checked by reading, not compiled, not run on hardware
//
#include <hip/hip_runtime.h>
#include <hip/hip_bf16.h>

// ---------------------------------------------------------------------------
// TreeLSTM on a complete binary tree, N = 2^18 - 1, H = 256.
// Children of node i are rows 2i+1, 2i+2 => h_cat / c_sum are contiguous-row
// views; every GEMM is dense. Matrix math: v_wmma_f32_16x16x32_bf16.
// All GEMM operands are pre-staged as bf16 (x leaf rows, W/U fragment-order,
// h maintained as a bf16 shadow), so the hot loops are fully unrolled
// global_load_b128 with constant immediate offsets feeding WMMA directly.
// ---------------------------------------------------------------------------

typedef __attribute__((ext_vector_type(16))) __bf16 v16bf;
typedef __attribute__((ext_vector_type(8)))  float  v8f;

#define N_NODES   262143
#define N_LEAVES  131072
#define LEAF0     131071   // first leaf node index
#define X_SIZE    300
#define KPAD_W    320      // 300 padded to multiple of 32
#define H_SIZE    256

union AFrag { v16bf bf; unsigned u[8]; uint4 q[2]; };
union BFrag { v16bf bf; unsigned u[8]; uint4 q[2]; };

__device__ __forceinline__ unsigned short f32_to_bf16(float f) {
    union { float f; unsigned u; } v; v.f = f;
    unsigned u = v.u;
    u += 0x7FFFu + ((u >> 16) & 1u);   // round-to-nearest-even
    return (unsigned short)(u >> 16);
}

__device__ __forceinline__ float sigmoidf_(float x) {
    return 1.0f / (1.0f + __expf(-x));
}

// Fragment-order offset for a bf16 weight matrix [Ncols][Kpad] (B operand):
//   tile = n/16, ks = k/32, grp = (k/8)&1, nl = n%16, j = (k%8) + 8*((k/16)&1)
//   off  = ((tile*(Kpad/32) + ks)*2 + grp)*256 + nl*16 + j     (shorts)
__device__ __forceinline__ size_t frag_off(int n, int k, int kpad) {
    int tile = n >> 4, nl = n & 15;
    int ks = k >> 5, grp = (k >> 3) & 1, j = (k & 7) + (((k >> 4) & 1) << 3);
    return ((size_t)((tile * (kpad >> 5) + ks) * 2 + grp) << 8) + (nl << 4) + j;
}

// ---------------------------------------------------------------------------
// Prep (one streaming pass):
//   W (300x512 row-major) -> Wt bf16 fragment-order, K padded to 320
//   U (512x1024 row-major) -> Ut bf16 fragment-order
//   x leaf rows            -> xbf bf16 [N_LEAVES][KPAD_W] row-major, 0-padded
// ---------------------------------------------------------------------------
__global__ void prep_convert_kernel(const float* __restrict__ W,
                                    const float* __restrict__ U,
                                    const float* __restrict__ x,
                                    unsigned short* __restrict__ Wt,
                                    unsigned short* __restrict__ Ut,
                                    unsigned short* __restrict__ xbf) {
    size_t idx = (size_t)blockIdx.x * blockDim.x + threadIdx.x;
    const size_t wcount = 512 * KPAD_W;            // 163840
    const size_t ucount = 1024 * 512;              // 524288
    if (idx < wcount) {
        int n = (int)(idx / KPAD_W), k = (int)(idx % KPAD_W);
        float v = (k < X_SIZE) ? W[(size_t)k * 512 + n] : 0.0f;
        Wt[frag_off(n, k, KPAD_W)] = f32_to_bf16(v);
    } else if (idx < wcount + ucount) {
        size_t i2 = idx - wcount;
        int n = (int)(i2 / 512), k = (int)(i2 % 512);
        Ut[frag_off(n, k, 512)] = f32_to_bf16(U[(size_t)k * 1024 + n]);
    } else {
        size_t i3 = idx - wcount - ucount;
        if (i3 < (size_t)N_LEAVES * KPAD_W) {
            int row = (int)(i3 / KPAD_W), k = (int)(i3 % KPAD_W);
            float v = (k < X_SIZE) ? x[(size_t)(LEAF0 + row) * X_SIZE + k] : 0.0f;
            xbf[i3] = f32_to_bf16(v);
        }
    }
}

// ---------------------------------------------------------------------------
// Leaf init: node = LEAF0 + row :  g = tanh(x_row @ W + bW)
//            h[node] = g[:256] (f32 + bf16), c[node] = g[256:] (f32)
// Block: 128 threads (4 waves), tile = 16 rows x 512 cols; wave: 8 N-tiles.
// Fully unrolled: all loads at constant offsets from two base pointers.
// ---------------------------------------------------------------------------
__global__ __launch_bounds__(128) void leaf_init_kernel(
        const unsigned short* __restrict__ xbf,  // [N_LEAVES][KPAD_W] bf16
        const unsigned short* __restrict__ Wt,   // fragment-order bf16
        const float* __restrict__ bW,            // [512]
        float* __restrict__ hout,                // [N_NODES][256] f32
        unsigned short* __restrict__ hbf,        // [N_NODES][256] bf16
        float* __restrict__ cbuf)                // [N_NODES][256] f32
{
    const int lane = threadIdx.x & 31;
    const int wave = threadIdx.x >> 5;     // 0..3
    const int grp  = lane >> 4;            // K-half select
    const int leaf0 = blockIdx.x * 16;     // exact grid, no partial tiles

    const unsigned short* Aptr =
        xbf + (size_t)(leaf0 + (lane & 15)) * KPAD_W + grp * 8;
    const unsigned short* Bbase =
        Wt + wave * (8 * 5120) + grp * 256 + ((lane & 15) << 4);

    v8f acc[8];
    const v8f vzero = {0.f,0.f,0.f,0.f,0.f,0.f,0.f,0.f};
    #pragma unroll
    for (int t = 0; t < 8; ++t) acc[t] = vzero;

    #pragma unroll
    for (int ks = 0; ks < 10; ++ks) {              // K = 320 = 10 * 32
        AFrag af;
        af.q[0] = *reinterpret_cast<const uint4*>(Aptr + ks * 32);
        af.q[1] = *reinterpret_cast<const uint4*>(Aptr + ks * 32 + 16);
        #pragma unroll
        for (int t = 0; t < 8; ++t) {
            BFrag bfr;
            const uint4* bp = reinterpret_cast<const uint4*>(
                Bbase + t * 5120 + ks * 512);
            bfr.q[0] = bp[0]; bfr.q[1] = bp[1];
            acc[t] = __builtin_amdgcn_wmma_f32_16x16x32_bf16(
                false, af.bf, false, bfr.bf, (short)0, acc[t], false, false);
        }
    }

    // Epilogue: tanh, split h / c; h also stored bf16 for next level's GEMM
    #pragma unroll
    for (int t = 0; t < 8; ++t) {
        int col = wave * 128 + t * 16 + (lane & 15);   // 0..511
        float b = bW[col];
        #pragma unroll
        for (int v = 0; v < 8; ++v) {
            int node = LEAF0 + leaf0 + v + grp * 8;    // C tile: M = v + 8*grp
            float g = tanhf(acc[t][v] + b);
            if (col < H_SIZE) {
                hout[(size_t)node * H_SIZE + col] = g;
                hbf [(size_t)node * H_SIZE + col] = f32_to_bf16(g);
            } else {
                cbuf[(size_t)node * H_SIZE + (col - H_SIZE)] = g;
            }
        }
    }
}

// ---------------------------------------------------------------------------
// One tree level: nodes [a, a+m). A row r = bf16 h rows 2r+1,2r+2 = 512
// contiguous bf16 at hbf + (2r+1)*256. Block: 128 threads (4 waves);
// tile = 16 rows x 1024 cols. Wave w owns the same 64-column slice in all
// 4 gate quarters -> coupled i/o/u/f epilogue is wave-local.
// Fully unrolled: all loads at constant offsets from two base pointers.
// ---------------------------------------------------------------------------
__global__ __launch_bounds__(128) void tree_level_kernel(
        const unsigned short* __restrict__ hbf_in,  // bf16 h (children rows)
        float* __restrict__ hout,                   // f32 h out (d_out)
        unsigned short* __restrict__ hbf_out,       // bf16 h out
        float* __restrict__ cbuf,                   // f32 c (read + write)
        const unsigned short* __restrict__ Ut,      // fragment-order bf16
        const float* __restrict__ bU,               // [1024]
        int a, int m)
{
    const int lane = threadIdx.x & 31;
    const int wave = threadIdx.x >> 5;     // 0..3
    const int grp  = lane >> 4;
    const int tile_row0 = a + blockIdx.x * 16;

    int rA = tile_row0 + (lane & 15);
    if (rA >= a + m) rA = a + m - 1;       // clamp partial tiles (safe read)
    const unsigned short* Aptr =
        hbf_in + (size_t)(2 * rA + 1) * H_SIZE + grp * 8;   // 512 bf16 row
    const unsigned short* Bbase =
        Ut + wave * (4 * 8192) + grp * 256 + ((lane & 15) << 4);

    v8f acc[4][4];
    const v8f vzero = {0.f,0.f,0.f,0.f,0.f,0.f,0.f,0.f};
    #pragma unroll
    for (int q = 0; q < 4; ++q)
        #pragma unroll
        for (int t = 0; t < 4; ++t) acc[q][t] = vzero;

    #pragma unroll
    for (int ks = 0; ks < 16; ++ks) {              // K = 512 = 16 * 32
        AFrag af;
        af.q[0] = *reinterpret_cast<const uint4*>(Aptr + ks * 32);
        af.q[1] = *reinterpret_cast<const uint4*>(Aptr + ks * 32 + 16);
        #pragma unroll
        for (int q = 0; q < 4; ++q) {
            #pragma unroll
            for (int t = 0; t < 4; ++t) {
                BFrag bfr;
                const uint4* bp = reinterpret_cast<const uint4*>(
                    Bbase + q * 131072 + t * 8192 + ks * 512);
                bfr.q[0] = bp[0]; bfr.q[1] = bp[1];
                acc[q][t] = __builtin_amdgcn_wmma_f32_16x16x32_bf16(
                    false, af.bf, false, bfr.bf, (short)0, acc[q][t],
                    false, false);
            }
        }
    }

    // Epilogue: i,o,u,f gates -> c_new, h_new
    #pragma unroll
    for (int t = 0; t < 4; ++t) {
        int j = wave * 64 + t * 16 + (lane & 15);   // gate-local col 0..255
        float bi = bU[j];
        float bo = bU[256 + j];
        float bu = bU[512 + j];
        float bf = bU[768 + j];
        #pragma unroll
        for (int v = 0; v < 8; ++v) {
            int rloc = v + grp * 8;                 // C tile: M = v + 8*grp
            int r = tile_row0 + rloc;
            if (rloc + (int)blockIdx.x * 16 < m) {
                float gi = sigmoidf_(acc[0][t][v] + bi);
                float go = sigmoidf_(acc[1][t][v] + bo);
                float gu = tanhf    (acc[2][t][v] + bu);
                float gf = sigmoidf_(acc[3][t][v] + bf);
                float csum = cbuf[(size_t)(2*r+1) * H_SIZE + j]
                           + cbuf[(size_t)(2*r+2) * H_SIZE + j];
                float cn = gi * gu + gf * csum;
                float hn = go * tanhf(cn);
                hout   [(size_t)r * H_SIZE + j] = hn;
                hbf_out[(size_t)r * H_SIZE + j] = f32_to_bf16(hn);
                cbuf   [(size_t)r * H_SIZE + j] = cn;
            }
        }
    }
}

// ---------------------------------------------------------------------------
extern "C" void kernel_launch(void* const* d_in, const int* in_sizes, int n_in,
                              void* d_out, int out_size, void* d_ws, size_t ws_size,
                              hipStream_t stream) {
    const float* x  = (const float*)d_in[0];
    const float* W  = (const float*)d_in[1];
    const float* bW = (const float*)d_in[2];
    const float* U  = (const float*)d_in[3];
    const float* bU = (const float*)d_in[4];
    // d_in[5] (children) unused: complete binary tree -> children of i are 2i+1, 2i+2

    float* hout = (float*)d_out;                       // [N_NODES][256] f32
    char*  wsb  = (char*)d_ws;

    float*          cbuf = (float*)wsb;                                // 268 MB
    size_t off = (size_t)N_NODES * H_SIZE * sizeof(float);
    unsigned short* hbf  = (unsigned short*)(wsb + off);               // 134 MB
    off += (size_t)N_NODES * H_SIZE * sizeof(unsigned short);
    unsigned short* xbf  = (unsigned short*)(wsb + off);               // 84 MB
    off += (size_t)N_LEAVES * KPAD_W * sizeof(unsigned short);
    unsigned short* Wt   = (unsigned short*)(wsb + off);               // 320 KB
    off += (size_t)512 * KPAD_W * sizeof(unsigned short);
    unsigned short* Ut   = (unsigned short*)(wsb + off);               // 1 MB

    // 1) convert x leaf rows + swizzle weights into bf16 (one streaming pass)
    {
        size_t total = (size_t)512 * KPAD_W + (size_t)1024 * 512
                     + (size_t)N_LEAVES * KPAD_W;
        int blocks = (int)((total + 255) / 256);
        prep_convert_kernel<<<blocks, 256, 0, stream>>>(W, U, x, Wt, Ut, xbf);
    }

    // 2) leaf initialization (only leaves' init values are ever consumed)
    leaf_init_kernel<<<N_LEAVES / 16, 128, 0, stream>>>(xbf, Wt, bW,
                                                        hout, hbf, cbuf);

    // 3) levels d = 16 .. 0 (stream order gives the level dependency)
    for (int d = 16; d >= 0; --d) {
        int a = (1 << d) - 1;
        int m = 1 << d;
        int blocks = (m + 15) / 16;
        tree_level_kernel<<<blocks, 128, 0, stream>>>(hbf, hout, hbf, cbuf,
                                                      Ut, bU, a, m);
    }
}
